// WorldHeadTransitionMLP_44513041056258
// MI455X (gfx1250) — matrix operand order repaired and checked by reading, not compiled, not run on hardware
//
#include <hip/hip_runtime.h>

// ---------------- problem constants ----------------
#define B_ROWS   8192
#define EMB      128
#define HID      256
#define NODES    20001     // NODE_COUNT + 1
#define NWORLD   16
#define TM       32        // batch rows per tile
#define HEAD_NT  256       // node columns per head workgroup (8 waves x 32)

typedef __attribute__((ext_vector_type(16))) _Float16 v16h;
typedef __attribute__((ext_vector_type(8)))  float    v8f;

union V16U { v16h v; uint4 q[2]; };
union H8U  { _Float16 h[8]; uint4 q; };

__device__ __forceinline__ v8f wmma16(v16h a, v16h b, v8f c) {
  // (neg_a, A, neg_b, B, c_mod, C, reuse_a, reuse_b) -> v_wmma_f32_16x16x32_f16
  return __builtin_amdgcn_wmma_f32_16x16x32_f16(false, a, false, b, (short)0, c, false, false);
}

// A-operand (16x32 f16) from an f32 row-major row. ISA A layout:
//   e in [0,8):  K = k0 + 8*g + e      e in [8,16): K = k0 + 16 + 8*g + (e-8)
__device__ __forceinline__ v16h fragA_f32(const float* __restrict__ row, int k0, int g) {
  const float* p = row + k0 + 8 * g;
  float4 x0 = *(const float4*)(p);
  float4 x1 = *(const float4*)(p + 4);
  float4 x2 = *(const float4*)(p + 16);
  float4 x3 = *(const float4*)(p + 20);
  v16h v;
  v[0]  = (_Float16)x0.x; v[1]  = (_Float16)x0.y; v[2]  = (_Float16)x0.z; v[3]  = (_Float16)x0.w;
  v[4]  = (_Float16)x1.x; v[5]  = (_Float16)x1.y; v[6]  = (_Float16)x1.z; v[7]  = (_Float16)x1.w;
  v[8]  = (_Float16)x2.x; v[9]  = (_Float16)x2.y; v[10] = (_Float16)x2.z; v[11] = (_Float16)x2.w;
  v[12] = (_Float16)x3.x; v[13] = (_Float16)x3.y; v[14] = (_Float16)x3.z; v[15] = (_Float16)x3.w;
  return v;
}

// B-operand (32x16 f16) from an f32 row-major row of a [N][K] matrix.
// B layout (matches A): e in [0,8): K = k0 + 8*g + e; e in [8,16): K = k0 + 16 + 8*g + (e-8).
__device__ __forceinline__ v16h fragB_f32(const float* __restrict__ row, int k0, int g) {
  const float* p = row + k0 + 8 * g;
  float4 x0 = *(const float4*)(p);
  float4 x1 = *(const float4*)(p + 4);
  float4 x2 = *(const float4*)(p + 16);
  float4 x3 = *(const float4*)(p + 20);
  v16h v;
  v[0]  = (_Float16)x0.x; v[1]  = (_Float16)x0.y; v[2]  = (_Float16)x0.z; v[3]  = (_Float16)x0.w;
  v[4]  = (_Float16)x1.x; v[5]  = (_Float16)x1.y; v[6]  = (_Float16)x1.z; v[7]  = (_Float16)x1.w;
  v[8]  = (_Float16)x2.x; v[9]  = (_Float16)x2.y; v[10] = (_Float16)x2.z; v[11] = (_Float16)x2.w;
  v[12] = (_Float16)x3.x; v[13] = (_Float16)x3.y; v[14] = (_Float16)x3.z; v[15] = (_Float16)x3.w;
  return v;
}

// A-operand from f16 LDS row (two 16B ds_read_b128)
__device__ __forceinline__ v16h fragA_lds(const _Float16* row, int k0, int g) {
  V16U u;
  u.q[0] = *(const uint4*)(row + k0 + 8 * g);
  u.q[1] = *(const uint4*)(row + k0 + 16 + 8 * g);
  return u.v;
}

// B-operand from f16 LDS row (same K layout as A: two 8-half chunks)
__device__ __forceinline__ v16h fragB_lds(const _Float16* row, int k0, int g) {
  V16U u;
  u.q[0] = *(const uint4*)(row + k0 + 8 * g);
  u.q[1] = *(const uint4*)(row + k0 + 16 + 8 * g);
  return u.v;
}

// ---------------------------------------------------------------------------
// Kernel 1: shared MLP.  Computes h2 = relu(relu(X@W1^T+b1)@W2^T+b2) as f16.
// Transposed GEMMs: D'[j][b] = sum_k W[j,k] * Act[b,k]  (A = weight rows from
// global, B = activation tile in LDS) so each lane's 8 D-elements are 8
// consecutive output features -> one packed 16B store.
// ---------------------------------------------------------------------------
__global__ __launch_bounds__(256)
void mlp_kernel(const int* __restrict__ source, const int* __restrict__ mode,
                const float* __restrict__ se, const float* __restrict__ me,
                const float* __restrict__ W1, const float* __restrict__ b1,
                const float* __restrict__ W2, const float* __restrict__ b2,
                _Float16* __restrict__ h2out)
{
  __shared__ _Float16 Xs[TM][2 * EMB];   // 32 x 256 f16 = 16 KB
  __shared__ _Float16 H1s[TM][HID];      // 16 KB

  const int t  = threadIdx.x;
  const int wg = blockIdx.x;

  // ---- gather X tile: 8 threads per row, 32 floats each (se | me concat)
  {
    int r  = t >> 3;
    int c0 = (t & 7) * 32;
    int b  = wg * TM + r;
    const float* srow = se + (long)source[b] * EMB;
    const float* mrow = me + (long)mode[b] * EMB;
#pragma unroll
    for (int j = 0; j < 32; j += 4) {
      int col = c0 + j;
      const float* src = (col < EMB) ? (srow + col) : (mrow + (col - EMB));
      float4 f = *(const float4*)src;
      Xs[r][col]     = (_Float16)f.x;
      Xs[r][col + 1] = (_Float16)f.y;
      Xs[r][col + 2] = (_Float16)f.z;
      Xs[r][col + 3] = (_Float16)f.w;
    }
  }
  __syncthreads();

  const int lane = t & 31;
  const int ln   = lane & 15;
  const int g    = lane >> 4;
  const int wv   = t >> 5;          // wave id 0..7
  const int j0   = wv * 32;         // this wave's output-feature base

  // ---------------- layer 1 ----------------
  v8f acc[2][2] = {};
#pragma unroll
  for (int k0 = 0; k0 < 2 * EMB; k0 += 32) {
    v16h a0 = fragA_f32(W1 + (long)(j0 + ln) * (2 * EMB), k0, g);
    v16h a1 = fragA_f32(W1 + (long)(j0 + 16 + ln) * (2 * EMB), k0, g);
    v16h bb0 = fragB_lds(&Xs[ln][0], k0, g);
    v16h bb1 = fragB_lds(&Xs[16 + ln][0], k0, g);
    acc[0][0] = wmma16(a0, bb0, acc[0][0]);
    acc[0][1] = wmma16(a0, bb1, acc[0][1]);
    acc[1][0] = wmma16(a1, bb0, acc[1][0]);
    acc[1][1] = wmma16(a1, bb1, acc[1][1]);
  }
#pragma unroll
  for (int rb = 0; rb < 2; ++rb) {
    int jb = j0 + rb * 16 + 8 * g;          // j = jb + r, r=0..7
    float4 bi0 = *(const float4*)(b1 + jb);
    float4 bi1 = *(const float4*)(b1 + jb + 4);
    float bias[8] = {bi0.x, bi0.y, bi0.z, bi0.w, bi1.x, bi1.y, bi1.z, bi1.w};
#pragma unroll
    for (int cb = 0; cb < 2; ++cb) {
      int brow = cb * 16 + ln;
      H8U hu;
#pragma unroll
      for (int r = 0; r < 8; ++r)
        hu.h[r] = (_Float16)fmaxf(acc[rb][cb][r] + bias[r], 0.0f);
      *(uint4*)(&H1s[brow][jb]) = hu.q;     // 16B aligned ds_write_b128
    }
  }
  __syncthreads();

  // ---------------- layer 2 ----------------
  v8f acc2[2][2] = {};
#pragma unroll
  for (int k0 = 0; k0 < HID; k0 += 32) {
    v16h a0 = fragA_f32(W2 + (long)(j0 + ln) * HID, k0, g);
    v16h a1 = fragA_f32(W2 + (long)(j0 + 16 + ln) * HID, k0, g);
    v16h bb0 = fragB_lds(&H1s[ln][0], k0, g);
    v16h bb1 = fragB_lds(&H1s[16 + ln][0], k0, g);
    acc2[0][0] = wmma16(a0, bb0, acc2[0][0]);
    acc2[0][1] = wmma16(a0, bb1, acc2[0][1]);
    acc2[1][0] = wmma16(a1, bb0, acc2[1][0]);
    acc2[1][1] = wmma16(a1, bb1, acc2[1][1]);
  }
#pragma unroll
  for (int rb = 0; rb < 2; ++rb) {
    int jb = j0 + rb * 16 + 8 * g;
    float4 bi0 = *(const float4*)(b2 + jb);
    float4 bi1 = *(const float4*)(b2 + jb + 4);
    float bias[8] = {bi0.x, bi0.y, bi0.z, bi0.w, bi1.x, bi1.y, bi1.z, bi1.w};
#pragma unroll
    for (int cb = 0; cb < 2; ++cb) {
      int bglob = wg * TM + cb * 16 + ln;
      H8U hu;
#pragma unroll
      for (int r = 0; r < 8; ++r)
        hu.h[r] = (_Float16)fmaxf(acc2[rb][cb][r] + bias[r], 0.0f);
      *(uint4*)(h2out + (long)bglob * HID + jb) = hu.q;   // global_store_b128
    }
  }
}

// ---------------------------------------------------------------------------
// Bucketing: meta[] layout (ints):
//  [0..15] counts  [16..32] offsets  [33..48] cursor  [49..65] tileBase
//  [80..80+B) perm
// ---------------------------------------------------------------------------
__device__ __forceinline__ int world_of(int c) {
  int w = c - 1;
  return w < 0 ? 0 : (w > NWORLD - 1 ? NWORLD - 1 : w);
}

__global__ void zero_kernel(int* meta) {
  if (threadIdx.x < 16) meta[threadIdx.x] = 0;
}

__global__ void count_kernel(const int* __restrict__ ctx, int* meta) {
  int b = blockIdx.x * blockDim.x + threadIdx.x;
  if (b < B_ROWS) atomicAdd(&meta[world_of(ctx[b])], 1);
}

__global__ void scan_kernel(int* meta) {
  if (threadIdx.x == 0) {
    int off = 0, tb = 0;
    for (int w = 0; w < NWORLD; ++w) {
      int c = meta[w];
      meta[16 + w] = off;         // offsets
      meta[33 + w] = off;         // cursor
      meta[49 + w] = tb;          // tileBase
      off += c;
      tb  += (c + TM - 1) / TM;
    }
    meta[32] = off;               // == B_ROWS
    meta[65] = tb;                // total row tiles
  }
}

__global__ void scatter_kernel(const int* __restrict__ ctx, int* meta) {
  int b = blockIdx.x * blockDim.x + threadIdx.x;
  if (b < B_ROWS) {
    int pos = atomicAdd(&meta[33 + world_of(ctx[b])], 1);
    meta[80 + pos] = b;
  }
}

// ---------------------------------------------------------------------------
// Kernel 6: per-world head GEMM.  out[b, n] = h2[b] . headsW[w][n] + headsb[w][n]
// Workgroup = one 32-row tile of one world  x  256 node columns.
// ---------------------------------------------------------------------------
__global__ __launch_bounds__(256)
void head_kernel(const float* __restrict__ headsW, const float* __restrict__ headsb,
                 const _Float16* __restrict__ h2, const int* __restrict__ meta,
                 float* __restrict__ out)
{
  __shared__ _Float16 As[TM][HID];   // 16 KB h2 tile (f16)
  __shared__ int s_perm[TM];

  const int t    = threadIdx.x;
  const int tile = blockIdx.y;
  const int n0   = blockIdx.x * HEAD_NT;

  if (tile >= meta[65]) return;      // uniform exit

  // find owning world: last w with tileBase[w] <= tile
  int w = 0;
#pragma unroll
  for (int i = 1; i < NWORLD; ++i)
    if (tile >= meta[49 + i]) w = i;
  const int local     = tile - meta[49 + w];
  const int rowStart  = meta[16 + w] + local * TM;
  int rowsValid       = meta[16 + w + 1] - rowStart;
  if (rowsValid > TM) rowsValid = TM;
  const int* perm = meta + 80;

  if (t < TM) {
    int ri = t < rowsValid ? t : rowsValid - 1;   // clamp (dup row, masked later)
    s_perm[t] = perm[rowStart + ri];
  }
  __syncthreads();

  // stage A tile (h2 rows, already f16): 8 threads/row x 64B each
  {
    int r = t >> 3;
    const uint4* src = (const uint4*)(h2 + (long)s_perm[r] * HID + (t & 7) * 32);
    uint4* dst = (uint4*)&As[r][(t & 7) * 32];
    dst[0] = src[0]; dst[1] = src[1]; dst[2] = src[2]; dst[3] = src[3];
  }
  __syncthreads();

  const int lane = t & 31;
  const int ln   = lane & 15;
  const int g    = lane >> 4;
  const int wv   = t >> 5;

  const float* W  = headsW + (long)w * NODES * HID;
  const float* hb = headsb + (long)w * NODES;

  const int ncol0 = n0 + wv * 32 + ln;
  const int ncol1 = ncol0 + 16;
  const int nc0 = ncol0 < NODES ? ncol0 : NODES - 1;   // clamped load row
  const int nc1 = ncol1 < NODES ? ncol1 : NODES - 1;

  v8f acc[2][2] = {};   // [row-block of batch][node col-block]
#pragma unroll
  for (int k0 = 0; k0 < HID; k0 += 32) {
    v16h a0  = fragA_lds(&As[ln][0], k0, g);
    v16h a1  = fragA_lds(&As[16 + ln][0], k0, g);
    v16h bb0 = fragB_f32(W + (long)nc0 * HID, k0, g);
    v16h bb1 = fragB_f32(W + (long)nc1 * HID, k0, g);
    acc[0][0] = wmma16(a0, bb0, acc[0][0]);
    acc[1][0] = wmma16(a1, bb0, acc[1][0]);
    acc[0][1] = wmma16(a0, bb1, acc[0][1]);
    acc[1][1] = wmma16(a1, bb1, acc[1][1]);
  }

  const float bias0 = (ncol0 < NODES) ? hb[ncol0] : 0.0f;
  const float bias1 = (ncol1 < NODES) ? hb[ncol1] : 0.0f;

#pragma unroll
  for (int rb = 0; rb < 2; ++rb) {
#pragma unroll
    for (int r = 0; r < 8; ++r) {
      int rl = rb * 16 + 8 * g + r;          // local batch row (D: M = r + 8g)
      if (rl < rowsValid) {
        float* orow = out + (long)s_perm[rl] * NODES;
        if (ncol0 < NODES) orow[ncol0] = acc[rb][0][r] + bias0;
        if (ncol1 < NODES) orow[ncol1] = acc[rb][1][r] + bias1;
      }
    }
  }
}

// ---------------------------------------------------------------------------
extern "C" void kernel_launch(void* const* d_in, const int* in_sizes, int n_in,
                              void* d_out, int out_size, void* d_ws, size_t ws_size,
                              hipStream_t stream) {
  (void)in_sizes; (void)n_in; (void)out_size; (void)ws_size;
  const int*   source = (const int*)d_in[0];
  const int*   mode   = (const int*)d_in[1];
  const int*   ctx    = (const int*)d_in[2];
  const float* se     = (const float*)d_in[3];
  const float* me     = (const float*)d_in[4];
  const float* W1     = (const float*)d_in[5];
  const float* b1     = (const float*)d_in[6];
  const float* W2     = (const float*)d_in[7];
  const float* b2     = (const float*)d_in[8];
  const float* headsW = (const float*)d_in[9];
  const float* headsb = (const float*)d_in[10];
  float* out = (float*)d_out;

  _Float16* h2 = (_Float16*)d_ws;                                   // 4 MB
  int* meta = (int*)((char*)d_ws + (size_t)B_ROWS * HID * sizeof(_Float16));

  mlp_kernel<<<B_ROWS / TM, 256, 0, stream>>>(source, mode, se, me, W1, b1, W2, b2, h2);
  zero_kernel<<<1, 32, 0, stream>>>(meta);
  count_kernel<<<B_ROWS / 256, 256, 0, stream>>>(ctx, meta);
  scan_kernel<<<1, 32, 0, stream>>>(meta);
  scatter_kernel<<<B_ROWS / 256, 256, 0, stream>>>(ctx, meta);

  dim3 grid((NODES + HEAD_NT - 1) / HEAD_NT, B_ROWS / TM + NWORLD);
  head_kernel<<<grid, 256, 0, stream>>>(headsW, headsb, h2, meta, out);
}
